// aux_branch_35880156791480
// MI455X (gfx1250) — compile-verified, matrix operand order checked by reading, not run on hardware
//
#include <hip/hip_runtime.h>
#include <hip/hip_bf16.h>

typedef __bf16 bf16_t;
typedef __attribute__((ext_vector_type(16))) __bf16 v16bf;
typedef __attribute__((ext_vector_type(8)))  __bf16 v8bf;
typedef __attribute__((ext_vector_type(8)))  float  v8f;

#define N_PTS 131072
#define P_PTS 4096
#define G_GRP 16

__device__ __forceinline__ bf16_t f2bf(float f) {
    unsigned u = __builtin_bit_cast(unsigned, f);
    u += 0x7fffu + ((u >> 16) & 1u);           // round-to-nearest-even
    unsigned short h = (unsigned short)(u >> 16);
    return __builtin_bit_cast(bf16_t, h);
}

// A/B fragment for v_wmma_f32_16x16x32_bf16 (ISA 7.12.2 16-bit 16x32 layout):
// per lane: halves [k0 .. k0+7] and [k0+16 .. k0+23] of its row/column,
// where k0 already includes the +8 offset for lanes 16..31.
__device__ __forceinline__ v16bf load_frag_bf16(const bf16_t* __restrict__ rowbase, int k0) {
    v8bf lo = *(const v8bf*)(rowbase + k0);
    v8bf hi = *(const v8bf*)(rowbase + k0 + 16);
    return __builtin_shufflevector(lo, hi, 0,1,2,3,4,5,6,7,8,9,10,11,12,13,14,15);
}

__device__ __forceinline__ v16bf load_frag_f32(const float* __restrict__ rowbase, int k0) {
    v16bf r;
#pragma unroll
    for (int i = 0; i < 8; ++i) {
        r[i]     = f2bf(rowbase[k0 + i]);
        r[8 + i] = f2bf(rowbase[k0 + 16 + i]);
    }
    return r;
}

__device__ __forceinline__ v8f wmma_bf16(v16bf a, v16bf b, v8f c) {
    return __builtin_amdgcn_wmma_f32_16x16x32_bf16(false, a, false, b, (short)0, c, false, false);
}

// ---------------------------------------------------------------------------
// Prep: transpose + convert weights to bf16 [n][k] so B-fragments are
// contiguous-K 16B loads.  w2t: 512x128, pw1t: 256x512, pw2t: 128x256 halves.
// ---------------------------------------------------------------------------
__global__ void __launch_bounds__(256)
prep_kernel(const float* __restrict__ w2, const float* __restrict__ pw1,
            const float* __restrict__ pw2,
            bf16_t* __restrict__ w2t, bf16_t* __restrict__ pw1t,
            bf16_t* __restrict__ pw2t) {
    int idx = blockIdx.x * 256 + threadIdx.x;
    if (idx < 128 * 512) {
        int k = idx >> 9, n = idx & 511;
        w2t[n * 128 + k] = f2bf(w2[idx]);
        return;
    }
    int i2 = idx - 128 * 512;
    if (i2 < 512 * 256) {
        int k = i2 >> 8, n = i2 & 255;
        pw1t[n * 512 + k] = f2bf(pw1[i2]);
        return;
    }
    int i3 = i2 - 512 * 256;
    if (i3 < 256 * 128) {
        int k = i3 >> 7, n = i3 & 127;
        pw2t[n * 256 + k] = f2bf(pw2[i3]);
    }
}

// ---------------------------------------------------------------------------
// Phase 1: one block per (class, group): gather 256 points, 6->128 MLP in
// VALU, 128->512 in WMMA, fused max over the 256 rows -> fout[cg][512].
// ---------------------------------------------------------------------------
#define SH1 136   // padded row stride (halves) for the h buffer in LDS

__global__ void __launch_bounds__(256)
phase1_kernel(const float* __restrict__ points,
              const int* __restrict__ bidx, const int* __restrict__ pidx,
              const float* __restrict__ w1, const float* __restrict__ b1,
              const bf16_t* __restrict__ w2t, const float* __restrict__ b2,
              float* __restrict__ fout) {
    extern __shared__ __align__(16) char smem[];
    bf16_t* hbuf = (bf16_t*)smem;              // 256 x SH1 halves (~70 KB)

    const int tid = threadIdx.x;
    const int cg  = blockIdx.x;                // class*16 + group
    const int c   = cg >> 4;
    const int g   = cg & 15;

    // ---- gather + layer 1 (6 -> 128), one row per thread -------------------
    {
        const int p  = g * 256 + tid;
        const int bi = bidx[c * P_PTS + p];
        const int pi = pidx[c * P_PTS + p];
        float gg[6];
#pragma unroll
        for (int d = 0; d < 6; ++d)
            gg[d] = points[(bi * 6 + d) * N_PTS + pi];
        bf16_t* hrow = hbuf + tid * SH1;
#pragma unroll 4
        for (int j = 0; j < 128; ++j) {
            float a = b1[j];
#pragma unroll
            for (int d = 0; d < 6; ++d) a = fmaf(gg[d], w1[d * 128 + j], a);
            hrow[j] = f2bf(fmaxf(a, 0.f));
        }
    }
    __syncthreads();

    // ---- layer 2 (128 -> 512) via WMMA, fused group-max --------------------
    const int lane = tid & 31;
    const int w    = tid >> 5;                 // wave 0..7
    const int kh8  = (lane >> 4) * 8;          // +8 K-offset for upper lanes
    const int nrow = lane & 15;

#pragma unroll
    for (int i = 0; i < 4; ++i) {
        const int nt = w * 4 + i;              // column tile 0..31
        const int n0 = nt * 16;

        v16bf Bf[4];                           // B frags resident across row tiles
#pragma unroll
        for (int kc = 0; kc < 4; ++kc)
            Bf[kc] = load_frag_bf16(w2t + (n0 + nrow) * 128, kc * 32 + kh8);

        float mx[8];
#pragma unroll
        for (int r = 0; r < 8; ++r) mx[r] = -3.4e38f;

        for (int mt = 0; mt < 16; ++mt) {      // 16 row tiles = 256 rows
            v8f acc = {0.f, 0.f, 0.f, 0.f, 0.f, 0.f, 0.f, 0.f};
            const bf16_t* arow = hbuf + (mt * 16 + nrow) * SH1;
#pragma unroll
            for (int kc = 0; kc < 4; ++kc) {
                v16bf Af = load_frag_bf16(arow, kc * 32 + kh8);
                acc = wmma_bf16(Af, Bf[kc], acc);
            }
#pragma unroll
            for (int r = 0; r < 8; ++r) mx[r] = fmaxf(mx[r], acc[r]);
        }
        // fold 8 rows/lane, then combine the two half-row sets (xor 16)
        float m = mx[0];
#pragma unroll
        for (int r = 1; r < 8; ++r) m = fmaxf(m, mx[r]);
        m = fmaxf(m, __shfl_xor(m, 16, 32));
        if (lane < 16) {
            const int n = n0 + lane;
            // max(relu(x+b)) == relu(max(x)+b)
            fout[(size_t)cg * 512 + n] = fmaxf(m + b2[n], 0.f);
        }
    }
}

// ---------------------------------------------------------------------------
// Phase 2: one block (128 thr = 4 waves) per class.
// z1 = relu(BN(F@pw1+pb1)), z2 = relu(BN(z1@pw2+pb2)), row-norm, proto EMA.
// BN is over the 16-row G axis -> computed inside the D fragments.
// ---------------------------------------------------------------------------
#define SZ1 264   // z1 row stride (halves): 256 + 8 pad
#define SZ2 132   // z2 row stride (floats): 128 + 4 pad

__global__ void __launch_bounds__(128)
phase2_kernel(const float* __restrict__ F,          // (C*16) x 512
              const bf16_t* __restrict__ pw1t, const float* __restrict__ pb1,
              const float* __restrict__ g1,  const float* __restrict__ be1,
              const bf16_t* __restrict__ pw2t, const float* __restrict__ pb2,
              const float* __restrict__ g2,  const float* __restrict__ be2,
              const float* __restrict__ proto,
              float* __restrict__ feat_out, float* __restrict__ proto_out) {
    __shared__ __align__(16) bf16_t z1[16 * SZ1];
    __shared__ __align__(16) float  z2[16 * SZ2];
    __shared__ float scl[16];

    const int c    = blockIdx.x;
    const int tid  = threadIdx.x;
    const int lane = tid & 31;
    const int w    = tid >> 5;                 // wave 0..3
    const int kh8  = (lane >> 4) * 8;
    const int nrow = lane & 15;
    const float* Fbase = F + (size_t)c * 16 * 512;

    // ---- GEMM1: 16x512 @ 512x256, 4 col tiles per wave ---------------------
    {
        v8f acc[4];
#pragma unroll
        for (int i = 0; i < 4; ++i) acc[i] = (v8f){0.f,0.f,0.f,0.f,0.f,0.f,0.f,0.f};
        for (int kc = 0; kc < 16; ++kc) {
            v16bf Af = load_frag_f32(Fbase + nrow * 512, kc * 32 + kh8);
#pragma unroll
            for (int i = 0; i < 4; ++i) {
                const int n0 = (w * 4 + i) * 16;
                v16bf Bf = load_frag_bf16(pw1t + (n0 + nrow) * 512, kc * 32 + kh8);
                acc[i] = wmma_bf16(Af, Bf, acc[i]);
            }
        }
#pragma unroll
        for (int i = 0; i < 4; ++i) {
            const int j = (w * 4 + i) * 16 + nrow;   // output channel
            const float bias = pb1[j];
            float s = 0.f, q = 0.f;
#pragma unroll
            for (int r = 0; r < 8; ++r) {
                float x = acc[i][r] + bias;
                acc[i][r] = x; s += x; q += x * x;
            }
            s += __shfl_xor(s, 16, 32);
            q += __shfl_xor(q, 16, 32);
            const float mean = s * (1.f / 16.f);
            const float var  = q * (1.f / 16.f) - mean * mean;
            const float rs   = rsqrtf(var + 1e-5f) * g1[j];
            const float bb   = be1[j];
#pragma unroll
            for (int r = 0; r < 8; ++r) {
                float zv = fmaxf((acc[i][r] - mean) * rs + bb, 0.f);
                z1[(r + kh8) * SZ1 + j] = f2bf(zv);
            }
        }
    }
    __syncthreads();

    // ---- GEMM2: 16x256 @ 256x128, 2 col tiles per wave ---------------------
    {
        v8f acc[2];
#pragma unroll
        for (int i = 0; i < 2; ++i) acc[i] = (v8f){0.f,0.f,0.f,0.f,0.f,0.f,0.f,0.f};
        for (int kc = 0; kc < 8; ++kc) {
            v16bf Af = load_frag_bf16(z1 + nrow * SZ1, kc * 32 + kh8);
#pragma unroll
            for (int i = 0; i < 2; ++i) {
                const int n0 = (w * 2 + i) * 16;
                v16bf Bf = load_frag_bf16(pw2t + (n0 + nrow) * 256, kc * 32 + kh8);
                acc[i] = wmma_bf16(Af, Bf, acc[i]);
            }
        }
#pragma unroll
        for (int i = 0; i < 2; ++i) {
            const int j = (w * 2 + i) * 16 + nrow;
            const float bias = pb2[j];
            float s = 0.f, q = 0.f;
#pragma unroll
            for (int r = 0; r < 8; ++r) {
                float x = acc[i][r] + bias;
                acc[i][r] = x; s += x; q += x * x;
            }
            s += __shfl_xor(s, 16, 32);
            q += __shfl_xor(q, 16, 32);
            const float mean = s * (1.f / 16.f);
            const float var  = q * (1.f / 16.f) - mean * mean;
            const float rs   = rsqrtf(var + 1e-5f) * g2[j];
            const float bb   = be2[j];
#pragma unroll
            for (int r = 0; r < 8; ++r) {
                float zv = fmaxf((acc[i][r] - mean) * rs + bb, 0.f);
                z2[(r + kh8) * SZ2 + j] = zv;
            }
        }
    }
    __syncthreads();

    // ---- row L2 norms ------------------------------------------------------
    if (tid < 16) {
        float q = 0.f;
        for (int j = 0; j < 128; ++j) { float x = z2[tid * SZ2 + j]; q += x * x; }
        scl[tid] = 1.f / fmaxf(sqrtf(q), 1e-12f);
    }
    __syncthreads();

    // ---- feat: (16,129) with class id in last column -----------------------
    for (int idx = tid; idx < 16 * 129; idx += 128) {
        const int m = idx / 129, j = idx - m * 129;
        const float v = (j < 128) ? z2[m * SZ2 + j] * scl[m] : (float)(c + 1);
        feat_out[(size_t)(c * 16 + m) * 129 + j] = v;
    }
    // ---- proto EMA ---------------------------------------------------------
    {
        const int j = tid;   // 0..127
        float s = 0.f;
#pragma unroll
        for (int m = 0; m < 16; ++m) s += z2[m * SZ2 + j] * scl[m];
        proto_out[c * 128 + j] = 0.999f * proto[c * 128 + j] + 0.001f * (s * (1.f / 16.f));
    }
}

// ---------------------------------------------------------------------------
extern "C" void kernel_launch(void* const* d_in, const int* in_sizes, int n_in,
                              void* d_out, int out_size, void* d_ws, size_t ws_size,
                              hipStream_t stream) {
    (void)in_sizes; (void)n_in; (void)out_size; (void)ws_size;

    const float* points  = (const float*)d_in[0];
    const int*   bidx[3] = {(const int*)d_in[1], (const int*)d_in[3], (const int*)d_in[5]};
    const int*   pidx[3] = {(const int*)d_in[2], (const int*)d_in[4], (const int*)d_in[6]};
    const float* w1  = (const float*)d_in[7];
    const float* b1  = (const float*)d_in[8];
    const float* w2  = (const float*)d_in[9];
    const float* b2  = (const float*)d_in[10];
    const float* pw1 = (const float*)d_in[11];
    const float* pb1 = (const float*)d_in[12];
    const float* g1  = (const float*)d_in[13];
    const float* be1 = (const float*)d_in[14];
    const float* pw2 = (const float*)d_in[15];
    const float* pb2 = (const float*)d_in[16];
    const float* g2  = (const float*)d_in[17];
    const float* be2 = (const float*)d_in[18];
    const float* proto[3] = {(const float*)d_in[19], (const float*)d_in[20],
                             (const float*)d_in[21]};
    float* out = (float*)d_out;

    // workspace layout
    char*   ws   = (char*)d_ws;
    float*  Fbuf = (float*)ws;                           // 1248 x 512 f32
    bf16_t* w2t  = (bf16_t*)(ws + (size_t)1248 * 512 * 4);
    bf16_t* pw1t = w2t + 512 * 128;
    bf16_t* pw2t = pw1t + 256 * 512;

    prep_kernel<<<896, 256, 0, stream>>>(w2, pw1, pw2, w2t, pw1t, pw2t);

    static const int Cs[3]     = {13, 25, 40};
    static const int rowOff[3] = {0, 13 * 16, (13 + 25) * 16};   // 0, 208, 608

    const size_t lds1 = (size_t)256 * SH1 * sizeof(bf16_t);      // ~70 KB
    (void)hipFuncSetAttribute(reinterpret_cast<const void*>(phase1_kernel),
                              hipFuncAttributeMaxDynamicSharedMemorySize, (int)lds1);

    for (int l = 0; l < 3; ++l)
        phase1_kernel<<<Cs[l] * G_GRP, 256, lds1, stream>>>(
            points, bidx[l], pidx[l], w1, b1, w2t, b2,
            Fbuf + (size_t)rowOff[l] * 512);

    const size_t f0 = 0;
    const size_t f1 = f0 + (size_t)13 * 16 * 129;
    const size_t f2 = f1 + (size_t)25 * 16 * 129;
    const size_t p0 = f2 + (size_t)40 * 16 * 129;
    const size_t p1 = p0 + (size_t)13 * 128;
    const size_t p2 = p1 + (size_t)25 * 128;
    const size_t featOff[3]  = {f0, f1, f2};
    const size_t protoOff[3] = {p0, p1, p2};

    for (int l = 0; l < 3; ++l)
        phase2_kernel<<<Cs[l], 128, 0, stream>>>(
            Fbuf + (size_t)rowOff[l] * 512,
            pw1t, pb1, g1, be1, pw2t, pb2, g2, be2,
            proto[l], out + featOff[l], out + protoOff[l]);
}
